// SelfAttention_3478923509950
// MI455X (gfx1250) — compile-verified, matrix operand order checked by reading, not run on hardware
//
#include <hip/hip_runtime.h>
#include <math.h>

// ---------------------------------------------------------------------------
// Performer (FAVOR+) self-attention for gfx1250 (MI455X).
// bf16 WMMA pipeline + CDNA5 async-to-LDS and TDM tensor_load_to_lds paths.
// ---------------------------------------------------------------------------

typedef __attribute__((ext_vector_type(16))) __bf16 v16bf;
typedef __attribute__((ext_vector_type(8)))  float  v8f;
typedef __attribute__((ext_vector_type(4)))  unsigned int u32x4;
typedef __attribute__((ext_vector_type(8)))  unsigned int u32x8;

union ABFrag { v16bf v; unsigned int u[8]; };

__device__ __forceinline__ unsigned short f2bf(float f) {
    unsigned u = __float_as_uint(f);
    u += 0x7fffu + ((u >> 16) & 1u);          // round-to-nearest-even
    return (unsigned short)(u >> 16);
}
__device__ __forceinline__ float bf2f(unsigned short h) {
    return __uint_as_float(((unsigned)h) << 16);
}
// bf16 A/B fragment k-pair offset for pair p (0..7), wave32 half (0/1).
__device__ __forceinline__ int kpair(int p, int half) {
    return ((p < 4) ? (2 * p) : (16 + 2 * (p - 4))) + half * 8;
}
__device__ __forceinline__ v8f wmma_bf16(const ABFrag& a, const ABFrag& b, v8f c) {
    return __builtin_amdgcn_wmma_f32_16x16x32_bf16(false, a.v, false, b.v,
                                                   (short)0, c, false, false);
}
// order-preserving float<->uint mapping for atomicMax over signed floats
__device__ __forceinline__ unsigned fkey(float f) {
    unsigned u = __float_as_uint(f);
    return (u & 0x80000000u) ? ~u : (u | 0x80000000u);
}
__device__ __forceinline__ float keyf(unsigned k) {
    return __uint_as_float((k & 0x80000000u) ? (k & 0x7fffffffu) : ~k);
}

// LDS aperture keeps the allocation-relative byte offset in addr[31:0].
__device__ __forceinline__ unsigned lds_u32(const void* p) {
    return (unsigned)(uintptr_t)p;
}
// Per-lane async global->LDS 16B copy (tracked with ASYNCcnt).
__device__ __forceinline__ void async_load_b128(unsigned lds_dst, const void* gsrc) {
    asm volatile("global_load_async_to_lds_b128 %0, %1, off"
                 :: "v"(lds_dst), "v"(gsrc) : "memory");
}
__device__ __forceinline__ void wait_asynccnt0() {
    asm volatile("s_wait_asynccnt 0x0" ::: "memory");
}
// TDM: DMA a contiguous run of `nelem` bf16 elements from global to LDS.
// D# group0/group1 built per CDNA5 ISA (2-group form).
__device__ __forceinline__ void tdm_copy_1d(unsigned lds_dst, const void* gsrc,
                                            unsigned nelem) {
    unsigned long long ga = (unsigned long long)(uintptr_t)gsrc;
    u32x4 g0;
    g0[0] = 1u;                                                 // count=1 (user D#)
    g0[1] = lds_dst;                                            // lds_addr
    g0[2] = (unsigned)(ga & 0xffffffffu);                       // global_addr[31:0]
    g0[3] = ((unsigned)(ga >> 32) & 0x01ffffffu) | 0x80000000u; // addr[56:32] | type=2
    u32x8 g1;
    g1[0] = (1u << 16);                         // workgroup_mask=0, data_size=1 (2B)
    g1[1] = (nelem & 0xffffu) << 16;            // tensor_dim0[15:0]
    g1[2] = ((nelem >> 16) & 0xffffu)           // tensor_dim0[31:16]
            | (1u << 16);                       // tensor_dim1 = 1
    g1[3] = (nelem & 0xffffu) << 16;            // tile_dim0 (nelem <= 65535)
    g1[4] = 1u;                                 // tile_dim1=1, tile_dim2=0
    g1[5] = nelem;                              // tensor_dim0_stride[31:0]
    g1[6] = 0u;                                 // stride0[47:32], stride1[15:0]
    g1[7] = 0u;                                 // stride1[47:16]
    asm volatile("tensor_load_to_lds %0, %1" :: "s"(g0), "s"(g1) : "memory");
}

#define SEQ   4096
#define DIM   1024
#define HEADS 16
#define DHEAD 64
#define MFEAT 256
#define BATCH 4
#define BH    (BATCH * HEADS)      // 64
#define MROWS (BATCH * SEQ)        // 16384

// ---------------------------------------------------------------------------
__global__ void cast_bf16_kernel(const float* __restrict__ src,
                                 unsigned short* __restrict__ dst, int n) {
    int i = blockIdx.x * blockDim.x + threadIdx.x;
    int stride = gridDim.x * blockDim.x;
    for (; i < n; i += stride) dst[i] = f2bf(src[i]);
}

__global__ void init_ws_kernel(float* __restrict__ ctx_f,
                               float* __restrict__ kc,
                               unsigned int* __restrict__ gmax) {
    int i = blockIdx.x * blockDim.x + threadIdx.x;
    int stride = gridDim.x * blockDim.x;
    for (int j = i; j < BH * MFEAT * DHEAD; j += stride) ctx_f[j] = 0.f;
    for (int j = i; j < BH * MFEAT; j += stride) kc[j] = 0.f;
    if (i == 0) *gmax = 0u;   // maps to the most-negative float key
}

// ---------------------------------------------------------------------------
// QKV GEMM: [16384,1024] x [1024,1024] -> bf16, stored head-split [bh][n][d].
// A tile moved global->LDS with per-lane async b128 copies (ASYNCcnt).
// ---------------------------------------------------------------------------
__global__ __launch_bounds__(256)
void qkv_gemm_kernel(const unsigned short* __restrict__ xb,
                     const unsigned short* __restrict__ wqb,
                     const unsigned short* __restrict__ wkb,
                     const unsigned short* __restrict__ wvb,
                     unsigned short* __restrict__ qh,
                     unsigned short* __restrict__ kh,
                     unsigned short* __restrict__ vh) {
    __shared__ __align__(16) unsigned short lA [64][32];
    __shared__ __align__(16) unsigned short lBt[64][32];   // n-major (k pairs contiguous)
    const unsigned short* W = (blockIdx.z == 0) ? wqb : (blockIdx.z == 1) ? wkb : wvb;
    unsigned short*       D = (blockIdx.z == 0) ? qh  : (blockIdx.z == 1) ? kh  : vh;

    int tid  = threadIdx.x;
    int wave = tid >> 5, lane = tid & 31, half = lane >> 4, lrow = lane & 15;
    int mW = wave & 3, nW = wave >> 2;
    int blockM = blockIdx.x * 64, blockN = blockIdx.y * 64;

    int ar = tid >> 2, aseg = (tid & 3) * 8;   // A tile loader (16B per thread)
    int bk = tid & 31, bseg = (tid >> 5) * 8;  // B tile loader
    unsigned ldsA_dst = lds_u32(&lA[ar][aseg]);

    v8f acc0 = {}, acc1 = {};
    for (int k0 = 0; k0 < DIM; k0 += 32) {
        const unsigned short* ap_g = &xb[(size_t)(blockM + ar) * DIM + k0 + aseg];
        const unsigned short* bp_g = &W[(size_t)(k0 + bk) * DIM + blockN + bseg];
        uint4 bv = *(const uint4*)bp_g;
        if (k0 + 32 < DIM) __builtin_prefetch(bp_g + 32 * DIM, 0, 1);
        __syncthreads();                      // previous iteration's LDS readers done
        async_load_b128(ldsA_dst, ap_g);      // global -> LDS, no VGPR round-trip
        const unsigned short* bs = (const unsigned short*)&bv;
        #pragma unroll
        for (int i = 0; i < 8; ++i) lBt[bseg + i][bk] = bs[i];
        wait_asynccnt0();
        __syncthreads();

        ABFrag a, b0, b1;
        const unsigned short* ap  = &lA [mW * 16 + lrow][0];
        const unsigned short* bp0 = &lBt[nW * 32 + lrow][0];
        const unsigned short* bp1 = &lBt[nW * 32 + 16 + lrow][0];
        #pragma unroll
        for (int p = 0; p < 8; ++p) {
            a .u[p] = *(const unsigned int*)(ap  + kpair(p, half));
            b0.u[p] = *(const unsigned int*)(bp0 + kpair(p, half));
            b1.u[p] = *(const unsigned int*)(bp1 + kpair(p, half));
        }
        acc0 = wmma_bf16(a, b0, acc0);
        acc1 = wmma_bf16(a, b1, acc1);
    }
    #pragma unroll
    for (int j = 0; j < 8; ++j) {
        int row = blockM + mW * 16 + j + half * 8;
        int bI = row >> 12, nI = row & 4095;
        int gc0 = blockN + nW * 32 + lrow;
        int gc1 = gc0 + 16;
        D[(((size_t)(bI * HEADS + (gc0 >> 6))) * SEQ + nI) * DHEAD + (gc0 & 63)] = f2bf(acc0[j]);
        D[(((size_t)(bI * HEADS + (gc1 >> 6))) * SEQ + nI) * DHEAD + (gc1 & 63)] = f2bf(acc1[j]);
    }
}

// ---------------------------------------------------------------------------
// FAVOR+ feature map. 128 rows x 256 features per block, one bh per y-block.
// Tile operands DMA'd into LDS by the TDM (tensor_load_to_lds, TENSORcnt).
// MODE 0: query (row-max stabilizer, write q').
// MODE 1: key pass 1 (global max -> atomicMax on orderable uint).
// MODE 2: key pass 2 (exp with global stab, write k', accumulate k_cumsum
//         via LDS ds_add_f32 then one global atomic per column).
// ---------------------------------------------------------------------------
template <int MODE>
__global__ __launch_bounds__(256)
void feature_kernel(const unsigned short* __restrict__ data,   // [bh][4096][64]
                    const unsigned short* __restrict__ projb,  // [256][64]
                    unsigned short* __restrict__ outp,         // [bh][4096][256]
                    unsigned int* __restrict__ gmax,
                    float* __restrict__ kc) {                   // [bh][256]
    __shared__ __align__(16) unsigned short lA [128][64];   // 16 KB
    __shared__ __align__(16) unsigned short lBt[256][64];   // 32 KB (projT, n-major)
    __shared__ float lkc[256];
    __shared__ float lwmax[8];

    const float normz = 0.35355339059327373f;   // 64^-0.25
    const float ratio = 0.0625f;                // 256^-0.5

    int tid  = threadIdx.x;
    int wave = tid >> 5, lane = tid & 31, half = lane >> 4, lrow = lane & 15;
    int bh = blockIdx.y;
    int blockRow = blockIdx.x * 128;

    if (wave == 0) {   // wave-uniform branch; TDM issues once per wave
        tdm_copy_1d(lds_u32(&lA[0][0]),
                    &data[((size_t)bh * SEQ + blockRow) * DHEAD], 128 * 64);
        tdm_copy_1d(lds_u32(&lBt[0][0]), projb, 256 * 64);
        __builtin_amdgcn_s_wait_tensorcnt(0);
    }
    if (MODE == 2) lkc[tid] = 0.f;
    __syncthreads();

    // ||x||^2 * 0.5 * normz^2 for this wave's 16 rows (lanes 0..15)
    float diagv = 0.f;
    if (lane < 16) {
        const unsigned short* rp = &lA[wave * 16 + lane][0];
        float s = 0.f;
        #pragma unroll 8
        for (int i = 0; i < 64; ++i) { float x = bf2f(rp[i]); s += x * x; }
        diagv = s * 0.5f * 0.125f;
    }

    v8f zero = {};
    v8f acc[16];
    #pragma unroll
    for (int t = 0; t < 16; ++t) acc[t] = zero;

    const unsigned short* ap = &lA[wave * 16 + lrow][0];
    #pragma unroll
    for (int kk = 0; kk < 2; ++kk) {
        int k0 = kk * 32;
        ABFrag a;
        #pragma unroll
        for (int p = 0; p < 8; ++p)
            a.u[p] = *(const unsigned int*)(ap + k0 + kpair(p, half));
        #pragma unroll
        for (int t = 0; t < 16; ++t) {
            ABFrag b;
            const unsigned short* bp = &lBt[t * 16 + lrow][0];
            #pragma unroll
            for (int p = 0; p < 8; ++p)
                b.u[p] = *(const unsigned int*)(bp + k0 + kpair(p, half));
            acc[t] = wmma_bf16(a, b, acc[t]);
        }
    }

    if (MODE == 1) {                     // block max -> device max
        float m = -3.0e38f;
        #pragma unroll
        for (int t = 0; t < 16; ++t)
            #pragma unroll
            for (int j = 0; j < 8; ++j) m = fmaxf(m, acc[t][j]);
        for (int off = 16; off > 0; off >>= 1) m = fmaxf(m, __shfl_xor(m, off, 32));
        if (lane == 0) lwmax[wave] = m;
        __syncthreads();
        if (tid == 0) {
            float bm = lwmax[0];
            #pragma unroll
            for (int w = 1; w < 8; ++w) bm = fmaxf(bm, lwmax[w]);
            atomicMax(gmax, fkey(bm * normz));
        }
        return;
    }

    float stab[8];
    if (MODE == 0) {                     // per-row max over 256 features
        #pragma unroll
        for (int j = 0; j < 8; ++j) {
            float m = acc[0][j];
            #pragma unroll
            for (int t = 1; t < 16; ++t) m = fmaxf(m, acc[t][j]);
            m = fmaxf(m, __shfl_xor(m, 1, 32));
            m = fmaxf(m, __shfl_xor(m, 2, 32));
            m = fmaxf(m, __shfl_xor(m, 4, 32));
            m = fmaxf(m, __shfl_xor(m, 8, 32));
            stab[j] = m * normz;
        }
    } else {
        float g = keyf(*gmax);
        #pragma unroll
        for (int j = 0; j < 8; ++j) stab[j] = g;
    }

    #pragma unroll
    for (int t = 0; t < 16; ++t) {
        float csum = 0.f;
        #pragma unroll
        for (int j = 0; j < 8; ++j) {
            int lr = j + half * 8;
            float dg = __shfl(diagv, lr, 32);
            float val = ratio * (__expf(acc[t][j] * normz - dg - stab[j]) + 1e-4f);
            size_t row = (size_t)bh * SEQ + blockRow + wave * 16 + lr;
            outp[row * MFEAT + t * 16 + lrow] = f2bf(val);
            csum += val;
        }
        if (MODE == 2) {
            csum += __shfl_xor(csum, 16, 32);
            if (half == 0) atomicAdd(&lkc[t * 16 + lrow], csum);   // ds_add_f32
        }
    }
    if (MODE == 2) {
        __syncthreads();
        atomicAdd(&kc[bh * MFEAT + tid], lkc[tid]);                // global_atomic_add_f32
    }
}

// ---------------------------------------------------------------------------
// context[bh] = k'^T (4096x256)^T . v (4096x64) -> [256][64], f32 atomics.
// K split over 8 blocks of 512 rows. LDS-transposed tiles.
// ---------------------------------------------------------------------------
__global__ __launch_bounds__(256)
void context_kernel(const unsigned short* __restrict__ kp,   // [bh][4096][256]
                    const unsigned short* __restrict__ vh,   // [bh][4096][64]
                    float* __restrict__ ctx_f) {             // [bh][256][64]
    __shared__ __align__(16) unsigned short lAt[256][32];    // k'^T tile (feat-major)
    __shared__ __align__(16) unsigned short lBt[64][32];     // v^T tile (dd-major)
    int tid  = threadIdx.x;
    int wave = tid >> 5, lane = tid & 31, half = lane >> 4, lrow = lane & 15;
    int bh = blockIdx.x;
    int n0base = blockIdx.y * 512;

    int ar = tid & 31, aseg = (tid >> 5) * 32;   // k' loader: row n, 32 feats
    int br = tid & 31, bseg = (tid >> 5) * 8;    // v  loader: row n, 8 dds

    v8f zero = {};
    v8f acc[2][4];
    #pragma unroll
    for (int mi = 0; mi < 2; ++mi)
        #pragma unroll
        for (int ni = 0; ni < 4; ++ni) acc[mi][ni] = zero;

    for (int kk = 0; kk < 512; kk += 32) {
        int n0 = n0base + kk;
        const unsigned short* arow = &kp[((size_t)bh * SEQ + n0 + ar) * MFEAT + aseg];
        uint4 a0 = *(const uint4*)(arow +  0);
        uint4 a1 = *(const uint4*)(arow +  8);
        uint4 a2 = *(const uint4*)(arow + 16);
        uint4 a3 = *(const uint4*)(arow + 24);
        uint4 bv = *(const uint4*)&vh[((size_t)bh * SEQ + n0 + br) * DHEAD + bseg];
        __syncthreads();
        const unsigned short* s;
        s = (const unsigned short*)&a0;
        #pragma unroll
        for (int i = 0; i < 8; ++i) lAt[aseg +  0 + i][ar] = s[i];
        s = (const unsigned short*)&a1;
        #pragma unroll
        for (int i = 0; i < 8; ++i) lAt[aseg +  8 + i][ar] = s[i];
        s = (const unsigned short*)&a2;
        #pragma unroll
        for (int i = 0; i < 8; ++i) lAt[aseg + 16 + i][ar] = s[i];
        s = (const unsigned short*)&a3;
        #pragma unroll
        for (int i = 0; i < 8; ++i) lAt[aseg + 24 + i][ar] = s[i];
        s = (const unsigned short*)&bv;
        #pragma unroll
        for (int i = 0; i < 8; ++i) lBt[bseg + i][br] = s[i];
        __syncthreads();

        ABFrag bfr[4];
        #pragma unroll
        for (int ni = 0; ni < 4; ++ni) {
            const unsigned short* bp = &lBt[ni * 16 + lrow][0];
            #pragma unroll
            for (int p = 0; p < 8; ++p)
                bfr[ni].u[p] = *(const unsigned int*)(bp + kpair(p, half));
        }
        #pragma unroll
        for (int mi = 0; mi < 2; ++mi) {
            ABFrag a;
            const unsigned short* ap = &lAt[(wave * 2 + mi) * 16 + lrow][0];
            #pragma unroll
            for (int p = 0; p < 8; ++p)
                a.u[p] = *(const unsigned int*)(ap + kpair(p, half));
            #pragma unroll
            for (int ni = 0; ni < 4; ++ni)
                acc[mi][ni] = wmma_bf16(a, bfr[ni], acc[mi][ni]);
        }
    }
    #pragma unroll
    for (int mi = 0; mi < 2; ++mi)
        #pragma unroll
        for (int ni = 0; ni < 4; ++ni)
            #pragma unroll
            for (int j = 0; j < 8; ++j) {
                int mf = (wave * 2 + mi) * 16 + j + half * 8;
                int dd = ni * 16 + lrow;
                atomicAdd(&ctx_f[((size_t)bh * MFEAT + mf) * DHEAD + dd], acc[mi][ni][j]);
            }
}

// ctx_f [bh][256][64] f32 -> ctxT [bh][64][256] bf16
__global__ void ctx_cast_kernel(const float* __restrict__ ctx_f,
                                unsigned short* __restrict__ ctxT) {
    int i = blockIdx.x * blockDim.x + threadIdx.x;
    if (i >= BH * MFEAT * DHEAD) return;
    int bh = i >> 14, rem = i & 16383, mf = rem >> 6, dd = rem & 63;
    ctxT[((size_t)bh * DHEAD + dd) * MFEAT + mf] = f2bf(ctx_f[i]);
}

// ---------------------------------------------------------------------------
// out_head = d_inv * (q' . context). 64 rows x 64 cols per block.
// ctxT (32 KB) DMA'd into LDS with one TDM descriptor.
// ---------------------------------------------------------------------------
__global__ __launch_bounds__(256)
void outhead_kernel(const unsigned short* __restrict__ qp,    // [bh][4096][256]
                    const unsigned short* __restrict__ ctxT,  // [bh][64][256]
                    const float* __restrict__ kc,             // [bh][256]
                    unsigned short* __restrict__ oh) {        // [b][4096][1024]
    __shared__ __align__(16) unsigned short lBt[64][256];     // 32 KB
    __shared__ float lkc[256];
    __shared__ float ldinv[64];
    int tid  = threadIdx.x;
    int wave = tid >> 5, lane = tid & 31, half = lane >> 4, lrow = lane & 15;
    int bh = blockIdx.y;
    int blockRow = blockIdx.x * 64;

    if (wave == 0) {
        tdm_copy_1d(lds_u32(&lBt[0][0]), &ctxT[(size_t)bh * DHEAD * MFEAT],
                    DHEAD * MFEAT);
        __builtin_amdgcn_s_wait_tensorcnt(0);
    }
    lkc[tid] = kc[bh * MFEAT + tid];
    __syncthreads();

    if (tid < 64) {   // d_inv = 1 / (q'[row] . k_cumsum)
        const unsigned short* rp = &qp[((size_t)bh * SEQ + blockRow + tid) * MFEAT];
        float s = 0.f;
        #pragma unroll 8
        for (int i = 0; i < MFEAT; ++i) s += bf2f(rp[i]) * lkc[i];
        ldinv[tid] = 1.0f / s;
    }
    __syncthreads();

    int mt = wave & 3, nh = wave >> 2;
    v8f acc0 = {}, acc1 = {};
    const unsigned short* arow = &qp[((size_t)bh * SEQ + blockRow + mt * 16 + lrow) * MFEAT];
    for (int k0 = 0; k0 < MFEAT; k0 += 32) {
        ABFrag a, b0, b1;
        #pragma unroll
        for (int p = 0; p < 8; ++p) {
            a .u[p] = *(const unsigned int*)(arow + k0 + kpair(p, half));
            b0.u[p] = *(const unsigned int*)(&lBt[(nh * 2 + 0) * 16 + lrow][k0] + kpair(p, half));
            b1.u[p] = *(const unsigned int*)(&lBt[(nh * 2 + 1) * 16 + lrow][k0] + kpair(p, half));
        }
        acc0 = wmma_bf16(a, b0, acc0);
        acc1 = wmma_bf16(a, b1, acc1);
    }
    int b = bh >> 4, h = bh & 15;
    #pragma unroll
    for (int j = 0; j < 8; ++j) {
        int lr = mt * 16 + j + half * 8;
        float dinv = ldinv[lr];
        size_t base = ((size_t)b * SEQ + blockRow + lr) * DIM + h * DHEAD;
        oh[base + (nh * 2 + 0) * 16 + lrow] = f2bf(acc0[j] * dinv);
        oh[base + (nh * 2 + 1) * 16 + lrow] = f2bf(acc1[j] * dinv);
    }
}

// ---------------------------------------------------------------------------
// Final projection: oh [16384,1024] bf16 x wo bf16 + bo -> f32 d_out.
// A tile via per-lane async global->LDS b128 copies.
// ---------------------------------------------------------------------------
__global__ __launch_bounds__(256)
void out_gemm_kernel(const unsigned short* __restrict__ oh,
                     const unsigned short* __restrict__ wob,
                     const float* __restrict__ bo,
                     float* __restrict__ out) {
    __shared__ __align__(16) unsigned short lA [64][32];
    __shared__ __align__(16) unsigned short lBt[64][32];
    int tid  = threadIdx.x;
    int wave = tid >> 5, lane = tid & 31, half = lane >> 4, lrow = lane & 15;
    int mW = wave & 3, nW = wave >> 2;
    int blockM = blockIdx.x * 64, blockN = blockIdx.y * 64;
    int ar = tid >> 2, aseg = (tid & 3) * 8;
    int bk = tid & 31, bseg = (tid >> 5) * 8;
    unsigned ldsA_dst = lds_u32(&lA[ar][aseg]);

    v8f acc0 = {}, acc1 = {};
    for (int k0 = 0; k0 < DIM; k0 += 32) {
        const unsigned short* ap_g = &oh [(size_t)(blockM + ar) * DIM + k0 + aseg];
        const unsigned short* bp_g = &wob[(size_t)(k0 + bk) * DIM + blockN + bseg];
        uint4 bv = *(const uint4*)bp_g;
        if (k0 + 32 < DIM) __builtin_prefetch(bp_g + 32 * DIM, 0, 1);
        __syncthreads();
        async_load_b128(ldsA_dst, ap_g);
        const unsigned short* bs = (const unsigned short*)&bv;
        #pragma unroll
        for (int i = 0; i < 8; ++i) lBt[bseg + i][bk] = bs[i];
        wait_asynccnt0();
        __syncthreads();

        ABFrag a, b0, b1;
        const unsigned short* ap  = &lA [mW * 16 + lrow][0];
        const unsigned short* bp0 = &lBt[nW * 32 + lrow][0];
        const unsigned short* bp1 = &lBt[nW * 32 + 16 + lrow][0];
        #pragma unroll
        for (int p = 0; p < 8; ++p) {
            a .u[p] = *(const unsigned int*)(ap  + kpair(p, half));
            b0.u[p] = *(const unsigned int*)(bp0 + kpair(p, half));
            b1.u[p] = *(const unsigned int*)(bp1 + kpair(p, half));
        }
        acc0 = wmma_bf16(a, b0, acc0);
        acc1 = wmma_bf16(a, b1, acc1);
    }
    #pragma unroll
    for (int j = 0; j < 8; ++j) {
        int row = blockM + mW * 16 + j + half * 8;
        int gc0 = blockN + nW * 32 + lrow;
        int gc1 = gc0 + 16;
        out[(size_t)row * DIM + gc0] = acc0[j] + bo[gc0];
        out[(size_t)row * DIM + gc1] = acc1[j] + bo[gc1];
    }
}

// ---------------------------------------------------------------------------
extern "C" void kernel_launch(void* const* d_in, const int* in_sizes, int n_in,
                              void* d_out, int out_size, void* d_ws, size_t ws_size,
                              hipStream_t stream) {
    const float* x    = (const float*)d_in[0];
    const float* wq   = (const float*)d_in[1];
    const float* wk   = (const float*)d_in[2];
    const float* wv   = (const float*)d_in[3];
    const float* wo   = (const float*)d_in[4];
    const float* bo   = (const float*)d_in[5];
    const float* proj = (const float*)d_in[6];
    float* out = (float*)d_out;

    char* ws = (char*)d_ws;
    size_t off = 0;
    auto take = [&](size_t bytes) -> char* {
        char* p = ws + off;
        off = (off + bytes + 255) & ~(size_t)255;
        return p;
    };
    const size_t nX  = (size_t)MROWS * DIM;       // 16.7M
    const size_t nW  = (size_t)DIM * DIM;         // 1.05M
    const size_t nQK = (size_t)BH * SEQ * DHEAD;  // 16.7M
    const size_t nFP = (size_t)BH * SEQ * MFEAT;  // 67.1M

    unsigned short* x_bf  = (unsigned short*)take(nX * 2);
    unsigned short* wq_bf = (unsigned short*)take(nW * 2);
    unsigned short* wk_bf = (unsigned short*)take(nW * 2);
    unsigned short* wv_bf = (unsigned short*)take(nW * 2);
    unsigned short* wo_bf = (unsigned short*)take(nW * 2);
    unsigned short* pj_bf = (unsigned short*)take((size_t)MFEAT * DHEAD * 2);
    unsigned short* q_bf  = (unsigned short*)take(nQK * 2);
    unsigned short* k_bf  = (unsigned short*)take(nQK * 2);
    unsigned short* v_bf  = (unsigned short*)take(nQK * 2);
    unsigned short* qp    = (unsigned short*)take(nFP * 2);
    unsigned short* kp    = (unsigned short*)take(nFP * 2);
    float*          kc    = (float*)take((size_t)BH * MFEAT * 4);
    unsigned int*   gmax  = (unsigned int*)take(256);
    float*          ctx_f = (float*)take((size_t)BH * MFEAT * DHEAD * 4);
    unsigned short* ctxT  = (unsigned short*)take((size_t)BH * MFEAT * DHEAD * 2);
    unsigned short* oh    = (unsigned short*)take(nX * 2);

    // 1) casts
    cast_bf16_kernel<<<2048, 256, 0, stream>>>(x,    x_bf,  (int)nX);
    cast_bf16_kernel<<<1024, 256, 0, stream>>>(wq,   wq_bf, (int)nW);
    cast_bf16_kernel<<<1024, 256, 0, stream>>>(wk,   wk_bf, (int)nW);
    cast_bf16_kernel<<<1024, 256, 0, stream>>>(wv,   wv_bf, (int)nW);
    cast_bf16_kernel<<<1024, 256, 0, stream>>>(wo,   wo_bf, (int)nW);
    cast_bf16_kernel<<<64,   256, 0, stream>>>(proj, pj_bf, MFEAT * DHEAD);

    // 2) zero accumulators / max sentinel
    init_ws_kernel<<<2048, 256, 0, stream>>>(ctx_f, kc, gmax);

    // 3) fused QKV projection (head-split bf16)
    qkv_gemm_kernel<<<dim3(MROWS / 64, DIM / 64, 3), 256, 0, stream>>>(
        x_bf, wq_bf, wk_bf, wv_bf, q_bf, k_bf, v_bf);

    // 4) FAVOR+ feature maps
    feature_kernel<0><<<dim3(SEQ / 128, BH), 256, 0, stream>>>(q_bf, pj_bf, qp, gmax, kc);
    feature_kernel<1><<<dim3(SEQ / 128, BH), 256, 0, stream>>>(k_bf, pj_bf, kp, gmax, kc);
    feature_kernel<2><<<dim3(SEQ / 128, BH), 256, 0, stream>>>(k_bf, pj_bf, kp, gmax, kc);

    // 5) context = k'^T v
    context_kernel<<<dim3(BH, 8), 256, 0, stream>>>(kp, v_bf, ctx_f);
    ctx_cast_kernel<<<(BH * MFEAT * DHEAD + 255) / 256, 256, 0, stream>>>(ctx_f, ctxT);

    // 6) per-head output with 1/(q'.k_cumsum) normalization
    outhead_kernel<<<dim3(SEQ / 64, BH), 256, 0, stream>>>(qp, ctxT, kc, oh);

    // 7) final projection + bias -> f32 output
    out_gemm_kernel<<<dim3(MROWS / 64, DIM / 64), 256, 0, stream>>>(oh, wo_bf, bo, out);

    (void)in_sizes; (void)n_in; (void)out_size; (void)ws_size;
}
